// Text2Motion_Transformer_75041668596238
// MI455X (gfx1250) — matrix operand order repaired and checked by reading, not compiled, not run on hardware
//
#include <hip/hip_runtime.h>
#include <hip/hip_bf16.h>
#include <math.h>

// ---------------------------------------------------------------------------
// Model dims (compile-time constants from the reference)
// ---------------------------------------------------------------------------
#define BB   2
#define LL   4
#define DD   512
#define HH   8
#define HDIM 64
#define TT   2048
#define TIDX 2047
#define SEM  50
#define ROWS (BB * TT)          // 4096
#define CLIPD 512

typedef __attribute__((ext_vector_type(16))) _Float16 v16h;
typedef __attribute__((ext_vector_type(8)))  float    v8f;
typedef __attribute__((ext_vector_type(4)))  int      v4i;

union FragH { v16h v; float4 f4[2]; };

// ---------------------------------------------------------------------------
// CDNA5 async global->LDS staging (ASYNCcnt path), with safe fallback.
// ---------------------------------------------------------------------------
#if defined(__has_builtin)
#if __has_builtin(__builtin_amdgcn_global_load_async_to_lds_b128) && \
    __has_builtin(__builtin_amdgcn_s_wait_asynccnt)
#define USE_ASYNC_LDS 1
#endif
#endif
#ifndef USE_ASYNC_LDS
#define USE_ASYNC_LDS 0
#endif

// copy 16 bytes (8 f16) global -> LDS (async, ASYNCcnt-tracked)
__device__ __forceinline__ void cp16(_Float16* dst, const _Float16* src) {
#if USE_ASYNC_LDS
  __builtin_amdgcn_global_load_async_to_lds_b128(
      (__attribute__((address_space(1))) v4i*)src,
      (__attribute__((address_space(3))) v4i*)dst, 0, 0);
#else
  *(float4*)dst = *(const float4*)src;
#endif
}
__device__ __forceinline__ void cp_join() {
#if USE_ASYNC_LDS
  __builtin_amdgcn_s_wait_asynccnt(0);
#endif
}

// WMMA D = A(16x32 f16) * B(32x16 f16) + C(16x16 f32)
__device__ __forceinline__ v8f wmma_f16(v16h a, v16h b, v8f c) {
  return __builtin_amdgcn_wmma_f32_16x16x32_f16(
      false, a, false, b, (short)0, c, false, false);
}

// A fragment (16x32 f16) from LDS tile stored [m][k], row stride stride_h
// halves. lane&15 = M; lanes 0-15 hold K 0..15, lanes 16-31 hold K 16..31
// (contiguous, matches B) -> one aligned 32B chunk.
__device__ __forceinline__ v16h load_a_frag(const _Float16* base, int stride_h,
                                            int lane) {
  FragH f;
  int m  = lane & 15;
  int ka = (lane >> 4) << 4;
  const _Float16* p = base + m * stride_h + ka;
  f.f4[0] = *(const float4*)(p);
  f.f4[1] = *(const float4*)(p + 8);
  return f.v;
}

// B fragment (32x16 f16) from LDS tile stored transposed [n][k].
// lane&15 = N; lanes 0-15 hold K 0..15, lanes 16-31 hold K 16..31
// (contiguous) -> one aligned 32B chunk.
__device__ __forceinline__ v16h load_b_frag(const _Float16* base, int stride_h,
                                            int lane) {
  FragH f;
  int n  = lane & 15;
  int kb = (lane >> 4) << 4;
  const _Float16* p = base + n * stride_h + kb;
  f.f4[0] = *(const float4*)(p);
  f.f4[1] = *(const float4*)(p + 8);
  return f.v;
}

// ---------------------------------------------------------------------------
// Embedding: x[b,t,:] = (t==0 ? clip@cond_w+cond_b : tok_emb[idx]) + pos_emb[t]
// ---------------------------------------------------------------------------
__global__ __launch_bounds__(256) void embed_kernel(
    const int* __restrict__ idx, const float* __restrict__ clip,
    const float* __restrict__ te0, const float* __restrict__ te1,
    const float* __restrict__ cw, const float* __restrict__ cb,
    const float* __restrict__ pe, float* __restrict__ x) {
  int bt = blockIdx.x;              // 0..4095
  int b  = bt >> 11;
  int t  = bt & (TT - 1);
  int tid = threadIdx.x;
  float* xr = x + (size_t)bt * DD;
  const float* per = pe + (size_t)t * DD;
  if (t == 0) {
    const float* cl = clip + (size_t)b * CLIPD;
    for (int d = tid; d < DD; d += 256) {
      float s = cb[d];
      for (int c = 0; c < CLIPD; ++c) s += cl[c] * cw[(size_t)c * DD + d];
      xr[d] = s + per[d];
    }
  } else {
    int j  = t - 1;
    int ix = idx[(size_t)b * TIDX + j];
    const float* e = (j < SEM ? te0 : te1) + (size_t)ix * DD;
    for (int d = tid; d < DD; d += 256) xr[d] = e[d] + per[d];
  }
}

// ---------------------------------------------------------------------------
// Weight convert + transpose: Wt[n][k] (f16) from W[k][n] (f32), 64x64 tiles
// ---------------------------------------------------------------------------
__global__ __launch_bounds__(256) void wcvt_kernel(
    const float* __restrict__ W, _Float16* __restrict__ Wt, int K, int N) {
  __shared__ alignas(16) _Float16 tbuf[64][72];
  int k0 = blockIdx.y * 64, n0 = blockIdx.x * 64;
  int tid = threadIdx.x;
  {
    int kk = tid >> 2, nn = (tid & 3) << 4;
    const float* src = W + (size_t)(k0 + kk) * N + n0 + nn;
    union { _Float16 h[16]; float4 f4[2]; } pk;
#pragma unroll
    for (int j = 0; j < 16; ++j) pk.h[j] = (_Float16)src[j];
    *(float4*)&tbuf[kk][nn]     = pk.f4[0];
    *(float4*)&tbuf[kk][nn + 8] = pk.f4[1];
  }
  __syncthreads();
  {
    int nn = tid >> 2, kk = (tid & 3) << 4;
    union { _Float16 h[16]; float4 f4[2]; } pk;
#pragma unroll
    for (int j = 0; j < 16; ++j) pk.h[j] = tbuf[kk + j][nn];
    _Float16* dst = Wt + (size_t)(n0 + nn) * K + k0 + kk;
    *(float4*)dst       = pk.f4[0];
    *(float4*)(dst + 8) = pk.f4[1];
  }
}

// ---------------------------------------------------------------------------
// LayerNorm over D=512, one block per row; f32 in, f16 out
// ---------------------------------------------------------------------------
__global__ __launch_bounds__(128) void ln_kernel(
    const float* __restrict__ x, const float* __restrict__ g,
    const float* __restrict__ b, _Float16* __restrict__ out) {
  __shared__ float red[128];
  int row = blockIdx.x, tid = threadIdx.x;
  const float* xr = x + (size_t)row * DD;
  float v[4], s = 0.f;
#pragma unroll
  for (int i = 0; i < 4; ++i) { v[i] = xr[tid + i * 128]; s += v[i]; }
  red[tid] = s; __syncthreads();
  for (int off = 64; off > 0; off >>= 1) {
    if (tid < off) red[tid] += red[tid + off];
    __syncthreads();
  }
  float mean = red[0] * (1.f / DD);
  __syncthreads();
  float s2 = 0.f;
#pragma unroll
  for (int i = 0; i < 4; ++i) { float d = v[i] - mean; s2 += d * d; }
  red[tid] = s2; __syncthreads();
  for (int off = 64; off > 0; off >>= 1) {
    if (tid < off) red[tid] += red[tid + off];
    __syncthreads();
  }
  float rstd = rsqrtf(red[0] * (1.f / DD) + 1e-5f);
  _Float16* orow = out + (size_t)row * DD;
#pragma unroll
  for (int i = 0; i < 4; ++i) {
    int d = tid + i * 128;
    orow[d] = (_Float16)((v[i] - mean) * rstd * g[d] + b[d]);
  }
}

// ---------------------------------------------------------------------------
// WMMA GEMM: C[MxN] = epi(A[MxK](f16) @ Wt[NxK]^T(f16) + bias (+resid f32))
// Block: 256 thr (8 waves). Macro-tile 64x64, K-block 64.
// Double-buffered LDS, single barrier per K-block; async loads for block i+1
// overlap the WMMAs of block i. Wave computes 16x32 (2 accs), A-frag reused.
// EPI : 0 = bias, 1 = bias + residual, 2 = bias + exact GELU
// OUTM: 0 = f32 row-major, 1 = f16 row-major,
//       2 = f16 transposed (B,N,T) for the V tensor (N=512, T=2048)
// ---------------------------------------------------------------------------
template <int EPI, int OUTM>
__global__ __launch_bounds__(256) void gemm_kernel(
    const _Float16* __restrict__ A, const _Float16* __restrict__ Wt,
    const float* __restrict__ bias, const float* __restrict__ resid,
    void* __restrict__ Cout, int M, int N, int K) {
  __shared__ alignas(16) _Float16 As[2][64][72];
  __shared__ alignas(16) _Float16 Bs[2][64][72];
  int tid = threadIdx.x, lane = tid & 31, w = tid >> 5;
  int m0 = blockIdx.y * 64, n0 = blockIdx.x * 64;
  int mw = (w & 3) << 4, nw = (w >> 2) << 5;
  v8f acc[2];
  acc[0] = (v8f){0.f, 0.f, 0.f, 0.f, 0.f, 0.f, 0.f, 0.f};
  acc[1] = (v8f){0.f, 0.f, 0.f, 0.f, 0.f, 0.f, 0.f, 0.f};

  int crow = tid >> 2, ccol = (tid & 3) << 4;   // 64 rows x 16 halves
  const _Float16* Ap = A  + (size_t)(m0 + crow) * K + ccol;
  const _Float16* Bp = Wt + (size_t)(n0 + crow) * K + ccol;

  auto stage = [&](int p, int kb) {
    cp16(&As[p][crow][ccol],     Ap + kb);
    cp16(&As[p][crow][ccol + 8], Ap + kb + 8);
    cp16(&Bs[p][crow][ccol],     Bp + kb);
    cp16(&Bs[p][crow][ccol + 8], Bp + kb + 8);
  };

  stage(0, 0);
  int p = 0;
  for (int kb = 0; kb < K; kb += 64, p ^= 1) {
    cp_join();              // this thread's loads for buffer p are done
    __syncthreads();        // everyone's loads done; prev reads of p^1 done
    if (kb + 64 < K) stage(p ^ 1, kb + 64);  // overlaps with compute below
#pragma unroll
    for (int ks = 0; ks < 64; ks += 32) {
      v16h af = load_a_frag(&As[p][mw][ks], 72, lane);
      v16h b0 = load_b_frag(&Bs[p][nw][ks], 72, lane);
      v16h b1 = load_b_frag(&Bs[p][nw + 16][ks], 72, lane);
      acc[0] = wmma_f16(af, b0, acc[0]);
      acc[1] = wmma_f16(af, b1, acc[1]);
    }
  }

  int mbase = m0 + mw + ((lane >> 4) << 3);
#pragma unroll
  for (int j = 0; j < 2; ++j) {
    int n = n0 + nw + (j << 4) + (lane & 15);
    float bn = bias[n];
    float vals[8];
#pragma unroll
    for (int r = 0; r < 8; ++r) {
      float vv = acc[j][r] + bn;
      if (EPI == 1) vv += resid[(size_t)(mbase + r) * N + n];
      if (EPI == 2) vv = 0.5f * vv * (1.0f + erff(vv * 0.70710678118654752f));
      vals[r] = vv;
    }
    if (OUTM == 0) {
      float* C = (float*)Cout;
#pragma unroll
      for (int r = 0; r < 8; ++r) C[(size_t)(mbase + r) * N + n] = vals[r];
    } else if (OUTM == 1) {
      _Float16* C = (_Float16*)Cout;
#pragma unroll
      for (int r = 0; r < 8; ++r)
        C[(size_t)(mbase + r) * N + n] = (_Float16)vals[r];
    } else {
      // transposed f16: C[b][n][t], t = mbase&2047 .. +7 contiguous
      union { _Float16 h[8]; float4 f4; } pk;
#pragma unroll
      for (int r = 0; r < 8; ++r) pk.h[r] = (_Float16)vals[r];
      _Float16* C = (_Float16*)Cout +
                    ((size_t)(mbase >> 11) * DD + n) * TT + (mbase & (TT - 1));
      *(float4*)C = pk.f4;
    }
  }
}

// ---------------------------------------------------------------------------
// Flash attention, causal; QK^T and PV on WMMA f16, softmax f32.
// Grid: (T/64, B*H). Block: 128 thr (4 waves); wave w owns 16 query rows.
// Double-buffered K/V, single barrier per key block; fully-masked blocks are
// skipped per-wave; cndmask only on diagonal tiles.
// q,k: f16 (B,T,H,HD) row stride D. vt: f16 (B, N=H*HD, T). y: f16 like q.
// ---------------------------------------------------------------------------
__global__ __launch_bounds__(128) void attn_kernel(
    const _Float16* __restrict__ q, const _Float16* __restrict__ k,
    const _Float16* __restrict__ vt, _Float16* __restrict__ y) {
  __shared__ alignas(16) _Float16 Qs[64][72];
  __shared__ alignas(16) _Float16 Ks[2][32][72];
  __shared__ alignas(16) _Float16 Vs[2][64][40];   // [hd][key]
  __shared__ alignas(16) _Float16 Ps[4][16][40];

  int qb0 = blockIdx.x * 64;
  int bh  = blockIdx.y;
  int b   = bh >> 3, hh = bh & 7;
  int tid = threadIdx.x, lane = tid & 31, w = tid >> 5;

  const _Float16* Qg = q + ((size_t)b * TT) * DD + hh * HDIM;
  const _Float16* Kg = k + ((size_t)b * TT) * DD + hh * HDIM;
  const _Float16* Vt = vt + ((size_t)b * DD + hh * HDIM) * TT;

  auto stageKV = [&](int p, int kb) {
    {  // K block (32 keys x 64 hd) -> Ks[p][key][hd]
      int key = tid >> 2, part = (tid & 3) << 4;
      const _Float16* src = Kg + (size_t)(kb + key) * DD + part;
      cp16(&Ks[p][key][part], src);
      cp16(&Ks[p][key][part + 8], src + 8);
    }
    {  // V^T block (64 hd x 32 keys) -> Vs[p][hd][key]
      int hd = tid >> 1, part = (tid & 1) << 4;
      const _Float16* src = Vt + (size_t)hd * TT + kb + part;
      cp16(&Vs[p][hd][part], src);
      cp16(&Vs[p][hd][part + 8], src + 8);
    }
  };

  {  // 64x64 Q tile -> Qs
    int row = tid >> 1, hoff = (tid & 1) << 5;
    const _Float16* src = Qg + (size_t)(qb0 + row) * DD + hoff;
#pragma unroll
    for (int j = 0; j < 4; ++j) cp16(&Qs[row][hoff + 8 * j], src + 8 * j);
  }
  stageKV(0, 0);

  v16h qf0 = {}, qf1 = {};
  float rmax[8], rsum[8];
  v8f oacc[4];
#pragma unroll
  for (int r = 0; r < 8; ++r) { rmax[r] = -3.0e38f; rsum[r] = 0.f; }
#pragma unroll
  for (int ht = 0; ht < 4; ++ht)
    oacc[ht] = (v8f){0.f, 0.f, 0.f, 0.f, 0.f, 0.f, 0.f, 0.f};

  int tqmin = qb0 + (w << 4);        // wave's lowest query row
  int kend = qb0 + 64;
  int p = 0;
  for (int kb = 0; kb < kend; kb += 32, p ^= 1) {
    cp_join();
    __syncthreads();
    if (kb + 32 < kend) stageKV(p ^ 1, kb + 32);   // overlaps compute
    if (kb == 0) {
      qf0 = load_a_frag(&Qs[w << 4][0], 72, lane);   // hd 0..31
      qf1 = load_a_frag(&Qs[w << 4][32], 72, lane);  // hd 32..63
    }
    if (kb > tqmin + 15) continue;   // block fully masked for this wave

    // S = Q K^T (two 16x16 n-tiles), causal mask only on diagonal tiles
    float sv[2][8];
#pragma unroll
    for (int nt = 0; nt < 2; ++nt) {
      v8f s = {0.f, 0.f, 0.f, 0.f, 0.f, 0.f, 0.f, 0.f};
      v16h kf0 = load_b_frag(&Ks[p][nt << 4][0], 72, lane);
      v16h kf1 = load_b_frag(&Ks[p][nt << 4][32], 72, lane);
      s = wmma_f16(qf0, kf0, s);
      s = wmma_f16(qf1, kf1, s);
      if (kb + (nt << 4) + 15 > tqmin) {           // diagonal tile: mask
        int tk = kb + (nt << 4) + (lane & 15);
#pragma unroll
        for (int r = 0; r < 8; ++r) {
          int tq = tqmin + r + ((lane >> 4) << 3);
          sv[nt][r] = (tk <= tq) ? s[r] * 0.125f : -3.0e38f;
        }
      } else {                                     // interior: no mask
#pragma unroll
        for (int r = 0; r < 8; ++r) sv[nt][r] = s[r] * 0.125f;
      }
    }

    // Online softmax
    float corr[8];
#pragma unroll
    for (int r = 0; r < 8; ++r) {
      float m2 = fmaxf(sv[0][r], sv[1][r]);
      for (int off = 8; off > 0; off >>= 1)
        m2 = fmaxf(m2, __shfl_xor(m2, off, 32));
      float nm = fmaxf(rmax[r], m2);
      corr[r] = expf(rmax[r] - nm);
      rmax[r] = nm;
    }
#pragma unroll
    for (int nt = 0; nt < 2; ++nt) {
#pragma unroll
      for (int r = 0; r < 8; ++r) {
        float pp = expf(sv[nt][r] - rmax[r]);
        sv[nt][r] = pp;
        Ps[w][r + ((lane >> 4) << 3)][(nt << 4) + (lane & 15)] = (_Float16)pp;
      }
    }
#pragma unroll
    for (int r = 0; r < 8; ++r) {
      float ps = sv[0][r] + sv[1][r];
      for (int off = 8; off > 0; off >>= 1) ps += __shfl_xor(ps, off, 32);
      rsum[r] = rsum[r] * corr[r] + ps;
#pragma unroll
      for (int ht = 0; ht < 4; ++ht) oacc[ht][r] *= corr[r];
    }

    // O += P V
    v16h pf = load_a_frag(&Ps[w][0][0], 40, lane);
#pragma unroll
    for (int ht = 0; ht < 4; ++ht) {
      v16h vf = load_b_frag(&Vs[p][ht << 4][0], 40, lane);
      oacc[ht] = wmma_f16(pf, vf, oacc[ht]);
    }
  }

#pragma unroll
  for (int r = 0; r < 8; ++r) rsum[r] = 1.f / rsum[r];
#pragma unroll
  for (int ht = 0; ht < 4; ++ht) {
    int hd = (ht << 4) + (lane & 15);
#pragma unroll
    for (int r = 0; r < 8; ++r) {
      int tq = tqmin + r + ((lane >> 4) << 3);
      y[((size_t)(b * TT + tq)) * DD + hh * HDIM + hd] =
          (_Float16)(oacc[ht][r] * rsum[r]);
    }
  }
}

// ---------------------------------------------------------------------------
// Host orchestration
// ---------------------------------------------------------------------------
extern "C" void kernel_launch(void* const* d_in, const int* in_sizes, int n_in,
                              void* d_out, int out_size, void* d_ws,
                              size_t ws_size, hipStream_t stream) {
  const int*   idx  = (const int*)d_in[0];
  const float* clip = (const float*)d_in[1];
  const float* te0  = (const float*)d_in[2];
  const float* te1  = (const float*)d_in[3];
  const float* cw   = (const float*)d_in[4];
  const float* cb   = (const float*)d_in[5];
  const float* pe   = (const float*)d_in[6];
  const float* ln1g = (const float*)d_in[7];
  const float* ln1b = (const float*)d_in[8];
  const float* wq   = (const float*)d_in[9];
  const float* bq   = (const float*)d_in[10];
  const float* wk   = (const float*)d_in[11];
  const float* bk   = (const float*)d_in[12];
  const float* wv   = (const float*)d_in[13];
  const float* bv   = (const float*)d_in[14];
  const float* wo   = (const float*)d_in[15];
  const float* bo   = (const float*)d_in[16];
  const float* ln2g = (const float*)d_in[17];
  const float* ln2b = (const float*)d_in[18];
  const float* w1   = (const float*)d_in[19];
  const float* b1   = (const float*)d_in[20];
  const float* w2   = (const float*)d_in[21];
  const float* b2   = (const float*)d_in[22];
  (void)in_sizes; (void)n_in; (void)ws_size; (void)out_size;

  const size_t RD = (size_t)ROWS * DD;          // 2M elements
  char* wsb = (char*)d_ws;
  float*    x   = (float*)wsb;                  // 8 MB f32 residual
  _Float16* h   = (_Float16*)(wsb + RD * 4);    // 4 MB f16 LN out
  _Float16* big = (_Float16*)(wsb + RD * 6);    // 16 MB f16 region
  _Float16* qh  = big;
  _Float16* kh  = qh + RD;
  _Float16* vT  = kh + RD;                      // (B, D, T)
  _Float16* yh  = vT + RD;
  _Float16* hid = big;                          // 4096x2048 overlays q,k,v,y
  _Float16* wtq = (_Float16*)(wsb + RD * 6 + RD * 8);  // weights region
  _Float16* wtk = wtq + (size_t)DD * DD;
  _Float16* wtv = wtk + (size_t)DD * DD;
  _Float16* wto = wtv + (size_t)DD * DD;
  _Float16* wt1 = wto + (size_t)DD * DD;        // [4D][D]
  _Float16* wt2 = wt1 + (size_t)4 * DD * DD;    // [D][4D]

  embed_kernel<<<ROWS, 256, 0, stream>>>(idx, clip, te0, te1, cw, cb, pe, x);

  dim3 gw_s(DD / 64, DD / 64);                  // 512x512 weights
  dim3 gw_1(4 * DD / 64, DD / 64);              // K=512, N=2048
  dim3 gw_2(DD / 64, 4 * DD / 64);              // K=2048, N=512
  dim3 gqkv(DD / 64, ROWS / 64);                // (8, 64)
  dim3 gmlp1(4 * DD / 64, ROWS / 64);           // (32, 64)
  dim3 gattn(TT / 64, BB * HH);                 // (32, 16)

  for (int i = 0; i < LL; ++i) {
    const size_t oDD  = (size_t)i * DD * DD;
    const size_t oD4D = (size_t)i * DD * 4 * DD;
    wcvt_kernel<<<gw_s, 256, 0, stream>>>(wq + oDD, wtq, DD, DD);
    wcvt_kernel<<<gw_s, 256, 0, stream>>>(wk + oDD, wtk, DD, DD);
    wcvt_kernel<<<gw_s, 256, 0, stream>>>(wv + oDD, wtv, DD, DD);
    wcvt_kernel<<<gw_s, 256, 0, stream>>>(wo + oDD, wto, DD, DD);
    wcvt_kernel<<<gw_1, 256, 0, stream>>>(w1 + oD4D, wt1, DD, 4 * DD);
    wcvt_kernel<<<gw_2, 256, 0, stream>>>(w2 + oD4D, wt2, 4 * DD, DD);

    ln_kernel<<<ROWS, 128, 0, stream>>>(x, ln1g + i * DD, ln1b + i * DD, h);
    gemm_kernel<0, 1><<<gqkv, 256, 0, stream>>>(h, wtq, bq + i * DD, nullptr,
                                                qh, ROWS, DD, DD);
    gemm_kernel<0, 1><<<gqkv, 256, 0, stream>>>(h, wtk, bk + i * DD, nullptr,
                                                kh, ROWS, DD, DD);
    gemm_kernel<0, 2><<<gqkv, 256, 0, stream>>>(h, wtv, bv + i * DD, nullptr,
                                                vT, ROWS, DD, DD);
    attn_kernel<<<gattn, 128, 0, stream>>>(qh, kh, vT, yh);
    gemm_kernel<1, 0><<<gqkv, 256, 0, stream>>>(yh, wto, bo + i * DD, x, x,
                                                ROWS, DD, DD);
    ln_kernel<<<ROWS, 128, 0, stream>>>(x, ln2g + i * DD, ln2b + i * DD, h);
    gemm_kernel<2, 1><<<gmlp1, 256, 0, stream>>>(h, wt1, b1 + i * 4 * DD,
                                                 nullptr, hid, ROWS, 4 * DD,
                                                 DD);
    float* outp = (i == LL - 1) ? (float*)d_out : x;
    gemm_kernel<1, 0><<<gqkv, 256, 0, stream>>>(hid, wt2, b2 + i * DD, x,
                                                outp, ROWS, DD, 4 * DD);
  }
}
